// MultilayerGRU_3212635538201
// MI455X (gfx1250) — matrix operand order repaired and checked by reading, not compiled, not run on hardware
//
#include <hip/hip_runtime.h>

#define Hdim 1024
#define Bdim 64
#define Sdim 512
#define Ldim 3

typedef __attribute__((ext_vector_type(16))) __bf16 v16bf;
typedef __attribute__((ext_vector_type(8)))  __bf16 v8bf;
typedef __attribute__((ext_vector_type(8)))  float  v8f;

__device__ __forceinline__ unsigned short f2bfu(float f) {
    unsigned u = __float_as_uint(f);
    u += 0x7fffu + ((u >> 16) & 1u);               // round-to-nearest-even
    return (unsigned short)(u >> 16);
}
__device__ __forceinline__ __bf16 f2bf(float f) {
    unsigned short s = f2bfu(f);
    return __builtin_bit_cast(__bf16, s);
}

__device__ __forceinline__ float sigmoidf_(float x) {
    return 1.0f / (1.0f + __expf(-x));
}
__device__ __forceinline__ float tanhf_(float x) {
    float e = __expf(2.0f * x);
    return (e - 1.0f) / (e + 1.0f);
}

__device__ __forceinline__ v8f wmma_bf16(v16bf a, v16bf b, v8f c) {
    return __builtin_amdgcn_wmma_f32_16x16x32_bf16(
        false, a, false, b, (short)0, c, false, false);
}

// ---- fragment loads straight from global (no LDS, no barriers) ----

// A fragment 16x32 (ISA layout) from f32 row-major, converted in-register.
__device__ __forceinline__ v16bf load_a_frag_f32(const float* __restrict__ src,
                                                 int rowStride, int k0, int rowBase) {
    const int lane = threadIdx.x & 31;
    const int m = lane & 15;
    const int h = lane >> 4;
    const float* p = src + (size_t)(rowBase + m) * rowStride + k0 + h * 8;
    const float4 f0 = *(const float4*)(p);
    const float4 f1 = *(const float4*)(p + 4);
    const float4 f2 = *(const float4*)(p + 16);
    const float4 f3 = *(const float4*)(p + 20);
    union { v16bf v; __bf16 e[16]; } u;
    u.e[0]  = f2bf(f0.x); u.e[1]  = f2bf(f0.y); u.e[2]  = f2bf(f0.z); u.e[3]  = f2bf(f0.w);
    u.e[4]  = f2bf(f1.x); u.e[5]  = f2bf(f1.y); u.e[6]  = f2bf(f1.z); u.e[7]  = f2bf(f1.w);
    u.e[8]  = f2bf(f2.x); u.e[9]  = f2bf(f2.y); u.e[10] = f2bf(f2.z); u.e[11] = f2bf(f2.w);
    u.e[12] = f2bf(f3.x); u.e[13] = f2bf(f3.y); u.e[14] = f2bf(f3.z); u.e[15] = f2bf(f3.w);
    return u.v;
}

// A fragment 16x32 from bf16 row-major: lane<16 -> row m=lane, K {0..7,16..23};
// lane>=16 -> row m=lane-16, K {8..15,24..31}.  Two aligned 16B loads.
__device__ __forceinline__ v16bf load_a_frag_bf(const unsigned short* __restrict__ src,
                                                int rowStride, int k0, int rowBase) {
    const int lane = threadIdx.x & 31;
    const int m = lane & 15;
    const int h = lane >> 4;
    const unsigned short* p = src + (size_t)(rowBase + m) * rowStride + k0 + h * 8;
    union { v16bf v; v8bf q[2]; } u;
    u.q[0] = *(const v8bf*)(p);
    u.q[1] = *(const v8bf*)(p + 16);
    return u.v;
}

// B fragment 32x16 from PRE-TRANSPOSED weights Wt[n][k] (row-major, ld=H):
// lane<16 -> col n=lane, K {0..7,16..23}; lane>=16 -> col n=lane-16, K {8..15,24..31}.
__device__ __forceinline__ v16bf load_b_frag_g(const unsigned short* __restrict__ Wt,
                                               int n0, int nt, int k0) {
    const int lane = threadIdx.x & 31;
    const int col = n0 + nt * 16 + (lane & 15);
    const int koff = (lane >> 4) * 8;
    const unsigned short* p = Wt + (size_t)col * Hdim + k0 + koff;
    __builtin_prefetch((const void*)(p + 32), 0, 1);   // next K chunk -> global_prefetch_b8
    union { v16bf v; v8bf q[2]; } u;
    u.q[0] = *(const v8bf*)(p);
    u.q[1] = *(const v8bf*)(p + 16);
    return u.v;
}

// ---- phase 1: r,z gates.  rh = sigmoid(in@Wxr + h@Whr + br) * h ; zbuf = sigmoid(...) ----
// grid = H/32 column tiles, block = 128 (4 waves; wave w owns rows 16w..16w+15)
template <bool AF32>
__global__ void __launch_bounds__(128)
gru_phase1(const void* __restrict__ inl, int inlStride,
           const float* __restrict__ hl_f, const unsigned short* __restrict__ hl_bf,
           const unsigned short* __restrict__ Wxr, const unsigned short* __restrict__ Whr,
           const unsigned short* __restrict__ Wxz, const unsigned short* __restrict__ Whz,
           const float* __restrict__ br, const float* __restrict__ bz,
           unsigned short* __restrict__ rh_bf, float* __restrict__ zbuf) {
    const int n0 = blockIdx.x * 32;
    const int wave = threadIdx.x >> 5;
    const int rowBase = wave * 16;
    v8f accr[2] = {};
    v8f accz[2] = {};

    for (int k0 = 0; k0 < Hdim; k0 += 32) {
        v16bf a1;
        if constexpr (AF32) a1 = load_a_frag_f32((const float*)inl, inlStride, k0, rowBase);
        else                a1 = load_a_frag_bf((const unsigned short*)inl, inlStride, k0, rowBase);
        const v16bf a2 = load_a_frag_bf(hl_bf, Hdim, k0, rowBase);
        // batch all B fragment loads (distinct regs -> clause + partial waits)
        const v16bf bxr0 = load_b_frag_g(Wxr, n0, 0, k0);
        const v16bf bxz0 = load_b_frag_g(Wxz, n0, 0, k0);
        const v16bf bxr1 = load_b_frag_g(Wxr, n0, 1, k0);
        const v16bf bxz1 = load_b_frag_g(Wxz, n0, 1, k0);
        const v16bf bhr0 = load_b_frag_g(Whr, n0, 0, k0);
        const v16bf bhz0 = load_b_frag_g(Whz, n0, 0, k0);
        const v16bf bhr1 = load_b_frag_g(Whr, n0, 1, k0);
        const v16bf bhz1 = load_b_frag_g(Whz, n0, 1, k0);
        // 4 independent accumulator chains, dependent WMMAs 4 issues apart
        accr[0] = wmma_bf16(a1, bxr0, accr[0]);
        accz[0] = wmma_bf16(a1, bxz0, accz[0]);
        accr[1] = wmma_bf16(a1, bxr1, accr[1]);
        accz[1] = wmma_bf16(a1, bxz1, accz[1]);
        accr[0] = wmma_bf16(a2, bhr0, accr[0]);
        accz[0] = wmma_bf16(a2, bhz0, accz[0]);
        accr[1] = wmma_bf16(a2, bhr1, accr[1]);
        accz[1] = wmma_bf16(a2, bhz1, accz[1]);
    }

    const int lane = threadIdx.x & 31;
    const int nl = lane & 15;
    const int half = lane >> 4;
#pragma unroll
    for (int nt = 0; nt < 2; ++nt) {
        const int col = n0 + nt * 16 + nl;
        const float brv = br[col];
        const float bzv = bz[col];
#pragma unroll
        for (int i = 0; i < 8; ++i) {
            const int b = rowBase + half * 8 + i;   // C/D layout: VGPR i -> M=i (+8 for hi lanes)
            const float hv = hl_f[b * Hdim + col];
            const float r = sigmoidf_(accr[nt][i] + brv);
            const float z = sigmoidf_(accz[nt][i] + bzv);
            rh_bf[b * Hdim + col] = f2bfu(r * hv);
            zbuf[b * Hdim + col] = z;
        }
    }
}

// ---- phase 2: g gate + state update.  h' = h*z + (1-z)*tanh(in@Wxg + rh@Whg + bg) ----
template <bool AF32>
__global__ void __launch_bounds__(128)
gru_phase2(const void* __restrict__ inl, int inlStride,
           const unsigned short* __restrict__ rh_bf,
           const unsigned short* __restrict__ Wxg, const unsigned short* __restrict__ Whg,
           const float* __restrict__ bg,
           const float* __restrict__ zbuf,
           float* __restrict__ hstate_l, unsigned short* __restrict__ hstate_l_bf) {
    const int n0 = blockIdx.x * 32;
    const int wave = threadIdx.x >> 5;
    const int rowBase = wave * 16;
    v8f accg[2] = {};

    for (int k0 = 0; k0 < Hdim; k0 += 32) {
        v16bf a1;
        if constexpr (AF32) a1 = load_a_frag_f32((const float*)inl, inlStride, k0, rowBase);
        else                a1 = load_a_frag_bf((const unsigned short*)inl, inlStride, k0, rowBase);
        const v16bf a2 = load_a_frag_bf(rh_bf, Hdim, k0, rowBase);
        const v16bf bxg0 = load_b_frag_g(Wxg, n0, 0, k0);
        const v16bf bxg1 = load_b_frag_g(Wxg, n0, 1, k0);
        const v16bf bhg0 = load_b_frag_g(Whg, n0, 0, k0);
        const v16bf bhg1 = load_b_frag_g(Whg, n0, 1, k0);
        accg[0] = wmma_bf16(a1, bxg0, accg[0]);
        accg[1] = wmma_bf16(a1, bxg1, accg[1]);
        accg[0] = wmma_bf16(a2, bhg0, accg[0]);
        accg[1] = wmma_bf16(a2, bhg1, accg[1]);
    }

    const int lane = threadIdx.x & 31;
    const int nl = lane & 15;
    const int half = lane >> 4;
#pragma unroll
    for (int nt = 0; nt < 2; ++nt) {
        const int col = n0 + nt * 16 + nl;
        const float bgv = bg[col];
#pragma unroll
        for (int i = 0; i < 8; ++i) {
            const int b = rowBase + half * 8 + i;
            const int idx = b * Hdim + col;
            const float g = tanhf_(accg[nt][i] + bgv);
            const float hv = hstate_l[idx];
            const float z = zbuf[idx];
            const float hn = hv * z + (1.0f - z) * g;
            hstate_l[idx] = hn;
            hstate_l_bf[idx] = f2bfu(hn);
        }
    }
}

// ---- output projection: y_t = h2 @ Wy + by ----
__global__ void __launch_bounds__(128)
gru_proj(const unsigned short* __restrict__ h2_bf,
         const unsigned short* __restrict__ Wy, const float* __restrict__ by,
         float* __restrict__ yout, int outRowStride) {
    const int n0 = blockIdx.x * 32;
    const int wave = threadIdx.x >> 5;
    const int rowBase = wave * 16;
    v8f acc[2] = {};

    for (int k0 = 0; k0 < Hdim; k0 += 32) {
        const v16bf a1 = load_a_frag_bf(h2_bf, Hdim, k0, rowBase);
        const v16bf by0 = load_b_frag_g(Wy, n0, 0, k0);
        const v16bf by1 = load_b_frag_g(Wy, n0, 1, k0);
        acc[0] = wmma_bf16(a1, by0, acc[0]);
        acc[1] = wmma_bf16(a1, by1, acc[1]);
    }

    const int lane = threadIdx.x & 31;
    const int nl = lane & 15;
    const int half = lane >> 4;
#pragma unroll
    for (int nt = 0; nt < 2; ++nt) {
        const int col = n0 + nt * 16 + nl;
        const float bv = by[col];
#pragma unroll
        for (int i = 0; i < 8; ++i) {
            const int b = rowBase + half * 8 + i;
            yout[(size_t)b * outRowStride + col] = acc[nt][i] + bv;
        }
    }
}

// ---- one-time weight convert + transpose: dst[n][k] = bf16(src[k][n]) ----
__global__ void __launch_bounds__(256)
cvt_transpose(const float* __restrict__ src, unsigned short* __restrict__ dst) {
    __shared__ float tile[32][33];
    const size_t moff = (size_t)blockIdx.z * Hdim * Hdim;
    const int kB = blockIdx.x * 32;
    const int nB = blockIdx.y * 32;
    const int tx = threadIdx.x & 31;
    const int ty = threadIdx.x >> 5;
#pragma unroll
    for (int i = 0; i < 32; i += 8)
        tile[ty + i][tx] = src[moff + (size_t)(kB + ty + i) * Hdim + nB + tx];
    __syncthreads();
#pragma unroll
    for (int i = 0; i < 32; i += 8) {
        dst[moff + (size_t)(nB + ty + i) * Hdim + kB + tx] = f2bfu(tile[tx][ty + i]);
    }
}

// h0: [B,L,H] -> hstate (f32 + bf16 shadow): [L,B,H]
__global__ void init_state(const float* __restrict__ h0, float* __restrict__ hstate,
                           unsigned short* __restrict__ hstate_bf) {
    int i = blockIdx.x * blockDim.x + threadIdx.x;
    if (i >= Ldim * Bdim * Hdim) return;
    int h = i % Hdim;
    int b = (i / Hdim) % Bdim;
    int l = i / (Hdim * Bdim);
    const float v = h0[(size_t)b * Ldim * Hdim + (size_t)l * Hdim + h];
    hstate[i] = v;
    hstate_bf[i] = f2bfu(v);
}

// hstate: [L,B,H] -> out hidden: [B,L,H]
__global__ void final_state(const float* __restrict__ hstate, float* __restrict__ outh) {
    int i = blockIdx.x * blockDim.x + threadIdx.x;
    if (i >= Ldim * Bdim * Hdim) return;
    int h = i % Hdim;
    int b = (i / Hdim) % Bdim;
    int l = i / (Hdim * Bdim);
    outh[(size_t)b * Ldim * Hdim + (size_t)l * Hdim + h] = hstate[i];
}

extern "C" void kernel_launch(void* const* d_in, const int* in_sizes, int n_in,
                              void* d_out, int out_size, void* d_ws, size_t ws_size,
                              hipStream_t stream) {
    (void)in_sizes; (void)n_in; (void)out_size; (void)ws_size;

    const float* x   = (const float*)d_in[0];   // [B,S,I]
    const float* h0  = (const float*)d_in[1];   // [B,L,H]
    const float* Wxz = (const float*)d_in[2];   // [L,H,H]
    const float* Wxr = (const float*)d_in[3];
    const float* Wxg = (const float*)d_in[4];
    const float* Whz = (const float*)d_in[5];
    const float* Whr = (const float*)d_in[6];
    const float* Whg = (const float*)d_in[7];
    const float* bz  = (const float*)d_in[8];   // [L,H]
    const float* br  = (const float*)d_in[9];
    const float* bg  = (const float*)d_in[10];
    const float* Wy  = (const float*)d_in[11];  // [H,O]
    const float* by  = (const float*)d_in[12];  // [O]

    float* out = (float*)d_out;                 // [B,S,O] ++ [B,L,H]

    // workspace: transposed bf16 weights (38 MB, L2-resident), f32 state, bf16 shadows
    const size_t M1 = (size_t)Hdim * Hdim;
    unsigned short* wbf = (unsigned short*)d_ws;
    unsigned short* Wxz_t = wbf + 0 * Ldim * M1;
    unsigned short* Wxr_t = wbf + 1 * Ldim * M1;
    unsigned short* Wxg_t = wbf + 2 * Ldim * M1;
    unsigned short* Whz_t = wbf + 3 * Ldim * M1;
    unsigned short* Whr_t = wbf + 4 * Ldim * M1;
    unsigned short* Whg_t = wbf + 5 * Ldim * M1;
    unsigned short* Wy_t  = wbf + 6 * Ldim * M1;
    float* fws    = (float*)(wbf + 6 * Ldim * M1 + M1);
    float* hstate = fws;                                   // [L,B,H] f32
    float* zbuf   = hstate + (size_t)Ldim * Bdim * Hdim;   // [B,H]   f32
    unsigned short* hstate_bf = (unsigned short*)(zbuf + (size_t)Bdim * Hdim);  // [L,B,H]
    unsigned short* rh_bf     = hstate_bf + (size_t)Ldim * Bdim * Hdim;         // [B,H]

    // 1) convert + transpose weights to bf16 (amortized over 512 steps)
    {
        const dim3 gG(Hdim / 32, Hdim / 32, Ldim);
        const dim3 g1(Hdim / 32, Hdim / 32, 1);
        cvt_transpose<<<gG, 256, 0, stream>>>(Wxz, Wxz_t);
        cvt_transpose<<<gG, 256, 0, stream>>>(Wxr, Wxr_t);
        cvt_transpose<<<gG, 256, 0, stream>>>(Wxg, Wxg_t);
        cvt_transpose<<<gG, 256, 0, stream>>>(Whz, Whz_t);
        cvt_transpose<<<gG, 256, 0, stream>>>(Whr, Whr_t);
        cvt_transpose<<<gG, 256, 0, stream>>>(Whg, Whg_t);
        cvt_transpose<<<g1, 256, 0, stream>>>(Wy, Wy_t);
    }

    // 2) init hidden state [L,B,H] (+ bf16 shadow)
    {
        const int n = Ldim * Bdim * Hdim;
        init_state<<<(n + 255) / 256, 256, 0, stream>>>(h0, hstate, hstate_bf);
    }

    // 3) sequential scan over time / layers
    const dim3 grid(Hdim / 32);   // 32 column tiles -> 128 waves/launch
    const dim3 block(128);        // 4 waves; wave w owns rows 16w..16w+15
    for (int t = 0; t < Sdim; ++t) {
        for (int l = 0; l < Ldim; ++l) {
            float* hl_f = hstate + (size_t)l * Bdim * Hdim;
            unsigned short* hl_bf = hstate_bf + (size_t)l * Bdim * Hdim;
            if (l == 0) {
                const void* inl = (const void*)(x + (size_t)t * Hdim);
                gru_phase1<true><<<grid, block, 0, stream>>>(
                    inl, Sdim * Hdim, hl_f, hl_bf,
                    Wxr_t + l * M1, Whr_t + l * M1, Wxz_t + l * M1, Whz_t + l * M1,
                    br + l * Hdim, bz + l * Hdim, rh_bf, zbuf);
                gru_phase2<true><<<grid, block, 0, stream>>>(
                    inl, Sdim * Hdim, rh_bf,
                    Wxg_t + l * M1, Whg_t + l * M1,
                    bg + l * Hdim, zbuf, hl_f, hl_bf);
            } else {
                const void* inl = (const void*)(hstate_bf + (size_t)(l - 1) * Bdim * Hdim);
                gru_phase1<false><<<grid, block, 0, stream>>>(
                    inl, Hdim, hl_f, hl_bf,
                    Wxr_t + l * M1, Whr_t + l * M1, Wxz_t + l * M1, Whz_t + l * M1,
                    br + l * Hdim, bz + l * Hdim, rh_bf, zbuf);
                gru_phase2<false><<<grid, block, 0, stream>>>(
                    inl, Hdim, rh_bf,
                    Wxg_t + l * M1, Whg_t + l * M1,
                    bg + l * Hdim, zbuf, hl_f, hl_bf);
            }
        }
        gru_proj<<<grid, block, 0, stream>>>(
            hstate_bf + (size_t)2 * Bdim * Hdim, Wy_t, by,
            out + (size_t)t * Hdim, Sdim * Hdim);
    }

    // 4) final hidden state [L,B,H] -> [B,L,H] at end of output buffer
    {
        const int n = Ldim * Bdim * Hdim;
        float* outh = out + (size_t)Bdim * Sdim * Hdim;
        final_state<<<(n + 255) / 256, 256, 0, stream>>>(hstate, outh);
    }
}
